// Attention_46566035423872
// MI455X (gfx1250) — matrix operand order repaired and checked by reading, not compiled, not run on hardware
//
#include <hip/hip_runtime.h>

// ---------------------------------------------------------------------------
// CDNA5 (gfx1250) fused causal attention with T5 relative-position bias.
// bf16 WMMA (v_wmma_f32_16x16x32_bf16) for all GEMMs, fp32 softmax.
// wave32; software-pipelined GEMM inner loop; prefetch hints.
// ---------------------------------------------------------------------------

typedef __attribute__((ext_vector_type(16))) __bf16 v16bf;
typedef __attribute__((ext_vector_type(8)))  float  v8f;
typedef __attribute__((ext_vector_type(4)))  float  v4f;
typedef __attribute__((ext_vector_type(4)))  __bf16 v4bf;

#define BATCH   2
#define NSEQ    2048
#define DMODEL  1024
#define HEADS   16
#define DIMH    64
#define INNER   1024   // HEADS*DIMH

__device__ __forceinline__ v8f vzero8() {
  v8f z = {0.f, 0.f, 0.f, 0.f, 0.f, 0.f, 0.f, 0.f};
  return z;
}

// A-matrix fragment (16x32 bf16), contiguous K order (matches B fragments):
// lane = half*16 + row; element j: K = half*16 + j
__device__ __forceinline__ v16bf load_a_frag(const __bf16* rowp, int half) {
  v16bf a;
#pragma unroll
  for (int j = 0; j < 16; ++j) {
    const int k = half * 16 + j;
    a[j] = rowp[k];
  }
  return a;
}

// ---------------------------------------------------------------------------
// vectorized fp32 -> bf16 (4 elements/thread)
__global__ void f32_to_bf16_v4(const v4f* __restrict__ in, v4bf* __restrict__ out, int n4) {
  const int i = blockIdx.x * blockDim.x + threadIdx.x;
  if (i >= n4) return;
  const v4f f = in[i];
  v4bf o;
#pragma unroll
  for (int j = 0; j < 4; ++j) o[j] = (__bf16)f[j];
  out[i] = o;
}

// fp32 [rows][cols] -> bf16 transposed [cols][rows] (so B-fragments load contiguously)
__global__ void f32_to_bf16_T(const float* __restrict__ in, __bf16* __restrict__ out,
                              int rows, int cols) {
  const int i = blockIdx.x * blockDim.x + threadIdx.x;
  if (i >= rows * cols) return;
  const int r = i / cols, c = i % cols;
  out[(size_t)c * rows + r] = (__bf16)in[i];
}

// vT[b][d][n] = kv[b][n][64+d]   (V columns of kv, transposed for PV B-fragments)
__global__ void transpose_v(const __bf16* __restrict__ kv, __bf16* __restrict__ vT) {
  const int i = blockIdx.x * blockDim.x + threadIdx.x;  // BATCH*64*NSEQ
  const int b = i >> 17;
  const int d = (i >> 11) & 63;
  const int n = i & (NSEQ - 1);
  vT[i] = kv[((size_t)(b * NSEQ + n)) * 128 + 64 + d];
}

// ---------------------------------------------------------------------------
// C[M x Ncols] = A[M x K](bf16, lda) @ BT[Ncols x K](bf16)
// 128 threads = 4 waves; each wave owns one 16x64 C tile (its own M tile).
// Software pipelined: next K-step's fragments load while current WMMAs issue.
// Cf!=nullptr: write fp32 (+bias).  else write bf16 to Cb.
__global__ void gemm_bf16(const __bf16* __restrict__ A, const __bf16* __restrict__ BT,
                          __bf16* __restrict__ Cb, float* __restrict__ Cf,
                          const float* __restrict__ bias,
                          int K, int lda, int ldc) {
  const int lane = threadIdx.x & 31;
  const int wave = threadIdx.x >> 5;
  const int row  = lane & 15;
  const int half = lane >> 4;
  const int m0 = (blockIdx.x * 4 + wave) * 16;
  const int n0 = blockIdx.y * 64;

  v8f acc[4];
#pragma unroll
  for (int t = 0; t < 4; ++t) acc[t] = vzero8();

  const __bf16* arow  = A + (size_t)(m0 + row) * lda;
  // B-frag: per-lane fixed col = n0 + t*16 + row, contiguous K chunk at half*16
  const __bf16* bbase = BT + (size_t)(n0 + row) * K + half * 16;

  // prologue: K-step 0
  v16bf a = load_a_frag(arow, half);
  v16bf b[4];
#pragma unroll
  for (int t = 0; t < 4; ++t)
    b[t] = *(const v16bf*)(bbase + (size_t)t * 16 * K);

  for (int k0 = 32; k0 < K; k0 += 32) {
    // issue next K-step's loads (no wait between these and the WMMAs below)
    const v16bf an = load_a_frag(arow + k0, half);
    v16bf bn[4];
#pragma unroll
    for (int t = 0; t < 4; ++t)
      bn[t] = *(const v16bf*)(bbase + (size_t)t * 16 * K + k0);

    __builtin_prefetch(arow + k0 + 96, 0, 3);  // global_prefetch ~3 steps ahead

#pragma unroll
    for (int t = 0; t < 4; ++t)
      acc[t] = __builtin_amdgcn_wmma_f32_16x16x32_bf16(
          false, a, false, b[t], (short)0, acc[t], false, false);

    a = an;
#pragma unroll
    for (int t = 0; t < 4; ++t) b[t] = bn[t];
  }
  // epilogue: last K-step
#pragma unroll
  for (int t = 0; t < 4; ++t)
    acc[t] = __builtin_amdgcn_wmma_f32_16x16x32_bf16(
        false, a, false, b[t], (short)0, acc[t], false, false);

#pragma unroll
  for (int t = 0; t < 4; ++t) {
    const int col = n0 + t * 16 + row;
#pragma unroll
    for (int r = 0; r < 8; ++r) {
      const int mr = m0 + r + half * 8;            // C layout: M = r + 8*half
      const float val = acc[t][r];
      const size_t idx = (size_t)mr * ldc + col;
      if (Cf) Cf[idx] = val + (bias ? bias[col] : 0.0f);
      else    Cb[idx] = (__bf16)val;
    }
  }
}

// ---------------------------------------------------------------------------
// T5 relative-position bucket for n = i - j >= 0 (causal side).
__device__ __forceinline__ int rp_bucket(int n) {
  if (n < 16) return n;
  const float inv_log8_x16 = 16.0f / 2.0794415416798357f;  // 16/log(128/16)
  int v = 16 + (int)(__logf((float)n * 0.0625f) * inv_log8_x16);
  return v < 31 ? v : 31;
}

// Fused flash attention: grid (N/16, HEADS, BATCH), 1 wave per 16-row tile.
__global__ void flash_attn(const __bf16* __restrict__ q, const __bf16* __restrict__ kv,
                           const __bf16* __restrict__ vT, const float* __restrict__ rel_emb,
                           __bf16* __restrict__ ao) {
  __shared__ __bf16 p_s[16][32];
  __shared__ float  bias_s[32];

  const int lane = threadIdx.x;
  const int row  = lane & 15;
  const int half = lane >> 4;
  const int mtile = blockIdx.x, h = blockIdx.y, b = blockIdx.z;
  const int i0 = mtile * 16;

  bias_s[lane] = rel_emb[lane * HEADS + h] * 8.0f;   // bias_scale = sqrt(dh) = 8
  __syncthreads();

  // Q tile 16x64 -> two A fragments (d 0..31, d 32..63)
  const __bf16* qrow = q + ((size_t)(b * NSEQ + i0 + row)) * INNER + h * DIMH;
  const v16bf qa0 = load_a_frag(qrow, half);
  const v16bf qa1 = load_a_frag(qrow + 32, half);

  float mrow[8], lrow[8];
  v8f O[4];
#pragma unroll
  for (int r = 0; r < 8; ++r) { mrow[r] = -1e30f; lrow[r] = 0.0f; }
#pragma unroll
  for (int d = 0; d < 4; ++d) O[d] = vzero8();

  const int kbcount = (16 * (mtile + 1) + 31) >> 5;    // 32-key blocks (causal)
  for (int kb = 0; kb < kbcount; ++kb) {
    // --- load ALL K and V fragments up front (V is independent of softmax) ---
    v16bf kf[2][2], vf[4];
#pragma unroll
    for (int t = 0; t < 2; ++t) {
      const int key = kb * 32 + t * 16 + row;          // B-frag: per-lane fixed key
      const __bf16* kp = kv + ((size_t)(b * NSEQ + key)) * 128 + half * 16;
      kf[t][0] = *(const v16bf*)kp;                    // d = half*16 + j
      kf[t][1] = *(const v16bf*)(kp + 32);             // d = 32 + half*16 + j
    }
#pragma unroll
    for (int d = 0; d < 4; ++d) {
      const __bf16* vp = vT + ((size_t)(b * DIMH + d * 16 + row)) * NSEQ
                       + kb * 32 + half * 16;          // contiguous in key
      vf[d] = *(const v16bf*)(vp);
    }
    if (kb + 1 < kbcount) {                            // prefetch next key block
      __builtin_prefetch(kv + ((size_t)(b * NSEQ + kb * 32 + 32 + lane)) * 128, 0, 3);
      __builtin_prefetch(vT + ((size_t)(b * DIMH + (lane & 31))) * NSEQ + kb * 32 + 32, 0, 3);
      __builtin_prefetch(vT + ((size_t)(b * DIMH + 32 + (lane & 31))) * NSEQ + kb * 32 + 32, 0, 3);
    }

    // --- S = Q K^T for 32 keys (two 16-col tiles) ---
    v8f acc[2];
#pragma unroll
    for (int t = 0; t < 2; ++t) {
      v8f z = vzero8();
      z      = __builtin_amdgcn_wmma_f32_16x16x32_bf16(false, qa0, false, kf[t][0],
                                                       (short)0, z, false, false);
      acc[t] = __builtin_amdgcn_wmma_f32_16x16x32_bf16(false, qa1, false, kf[t][1],
                                                       (short)0, z, false, false);
    }

    // --- online softmax (rows live in 16-lane halves; cols across lanes) ---
    const int jg0 = kb * 32 + row;
#pragma unroll
    for (int r = 0; r < 8; ++r) {
      const int ig = i0 + r + half * 8;
      float s0 = (jg0 <= ig) ? acc[0][r] * 0.125f + bias_s[rp_bucket(ig - jg0)] : -1e30f;
      const int jg1 = jg0 + 16;
      float s1 = (jg1 <= ig) ? acc[1][r] * 0.125f + bias_s[rp_bucket(ig - jg1)] : -1e30f;

      float rm = fmaxf(s0, s1);
#pragma unroll
      for (int msk = 1; msk < 16; msk <<= 1) rm = fmaxf(rm, __shfl_xor(rm, msk, 32));
      const float mnew = fmaxf(mrow[r], rm);
      const float corr = __expf(mrow[r] - mnew);
      mrow[r] = mnew;

      const float p0 = __expf(s0 - mnew);
      const float p1 = __expf(s1 - mnew);
      float rs = p0 + p1;
#pragma unroll
      for (int msk = 1; msk < 16; msk <<= 1) rs += __shfl_xor(rs, msk, 32);
      lrow[r] = lrow[r] * corr + rs;
#pragma unroll
      for (int d = 0; d < 4; ++d) O[d][r] *= corr;

      p_s[r + half * 8][row]      = (__bf16)p0;        // C-layout -> LDS
      p_s[r + half * 8][row + 16] = (__bf16)p1;
    }
    __syncthreads();
    const v16bf pa = load_a_frag(&p_s[row][0], half);  // LDS -> A-layout
    __syncthreads();

    // --- O += P V  (K=32 keys, four 16-wide dim tiles) ---
#pragma unroll
    for (int d = 0; d < 4; ++d)
      O[d] = __builtin_amdgcn_wmma_f32_16x16x32_bf16(false, pa, false, vf[d],
                                                     (short)0, O[d], false, false);
  }

  // --- normalize and store (B,N,h*dh layout) ---
#pragma unroll
  for (int d = 0; d < 4; ++d) {
#pragma unroll
    for (int r = 0; r < 8; ++r) {
      const int ig = i0 + r + half * 8;
      const int dd = d * 16 + row;
      const float val = O[d][r] / lrow[r];
      ao[((size_t)(b * NSEQ + ig)) * INNER + h * DIMH + dd] = (__bf16)val;
    }
  }
}

// ---------------------------------------------------------------------------
extern "C" void kernel_launch(void* const* d_in, const int* in_sizes, int n_in,
                              void* d_out, int out_size, void* d_ws, size_t ws_size,
                              hipStream_t stream) {
  (void)in_sizes; (void)n_in; (void)out_size; (void)ws_size;
  const float* x    = (const float*)d_in[0];
  const float* Wq   = (const float*)d_in[1];
  const float* Wkv  = (const float*)d_in[2];
  const float* Wout = (const float*)d_in[3];
  const float* bout = (const float*)d_in[4];
  const float* rel  = (const float*)d_in[5];

  char* w = (char*)d_ws;
  size_t o = 0;
  __bf16* x_bf   = (__bf16*)(w + o); o += (size_t)BATCH * NSEQ * DMODEL * 2;  // 8 MiB
  __bf16* WqT    = (__bf16*)(w + o); o += (size_t)DMODEL * INNER * 2;         // 2 MiB
  __bf16* WkvT   = (__bf16*)(w + o); o += (size_t)DMODEL * 128 * 2;           // 256 KiB
  __bf16* WoutT  = (__bf16*)(w + o); o += (size_t)INNER * DMODEL * 2;         // 2 MiB
  __bf16* q_bf   = (__bf16*)(w + o); o += (size_t)BATCH * NSEQ * INNER * 2;   // 8 MiB
  __bf16* kv_bf  = (__bf16*)(w + o); o += (size_t)BATCH * NSEQ * 128 * 2;     // 1 MiB
  __bf16* vT_bf  = (__bf16*)(w + o); o += (size_t)BATCH * DIMH * NSEQ * 2;    // 512 KiB
  __bf16* ao_bf  = (__bf16*)(w + o); o += (size_t)BATCH * NSEQ * INNER * 2;   // 8 MiB

  const int nx = BATCH * NSEQ * DMODEL;  // 4194304

  f32_to_bf16_v4<<<nx / 4 / 256, 256, 0, stream>>>((const v4f*)x, (v4bf*)x_bf, nx / 4);
  f32_to_bf16_T<<<(DMODEL * INNER) / 256, 256, 0, stream>>>(Wq,   WqT,   DMODEL, INNER);
  f32_to_bf16_T<<<(DMODEL * 128)   / 256, 256, 0, stream>>>(Wkv,  WkvT,  DMODEL, 128);
  f32_to_bf16_T<<<(INNER * DMODEL) / 256, 256, 0, stream>>>(Wout, WoutT, INNER,  DMODEL);

  // q = x @ Wq   (4096 x 1024, K=1024)
  gemm_bf16<<<dim3((BATCH * NSEQ) / 64, INNER / 64), 128, 0, stream>>>(
      x_bf, WqT, q_bf, nullptr, nullptr, DMODEL, DMODEL, INNER);
  // kv = x @ Wkv (4096 x 128, K=1024)
  gemm_bf16<<<dim3((BATCH * NSEQ) / 64, 128 / 64), 128, 0, stream>>>(
      x_bf, WkvT, kv_bf, nullptr, nullptr, DMODEL, DMODEL, 128);

  transpose_v<<<(BATCH * DIMH * NSEQ) / 256, 256, 0, stream>>>(kv_bf, vT_bf);

  flash_attn<<<dim3(NSEQ / 16, HEADS, BATCH), 32, 0, stream>>>(
      q_bf, kv_bf, vT_bf, rel, ao_bf);

  // out = attn_out @ Wout + bout  (fp32 output)
  gemm_bf16<<<dim3((BATCH * NSEQ) / 64, DMODEL / 64), 128, 0, stream>>>(
      ao_bf, WoutT, nullptr, (float*)d_out, bout, INNER, INNER, DMODEL);
}
